// Polynomial_481036337476
// MI455X (gfx1250) — matrix operand order repaired and checked by reading, not compiled, not run on hardware
//
#include <hip/hip_runtime.h>
#include <math.h>

// ---------------------------------------------------------------------------
// Polynomial feature kernel for MI455X (gfx1250, wave32, WMMA).
//   Xp = X @ P via bf16x3 split WMMA (fp32-comparable accuracy),
//   fused epilogue: bias | Xp | Xp^2 | Xp^3 | cross | sqrt(|Xp|+eps)
//
// Roofline: 306 MB total traffic -> ~13us floor @ 23.3 TB/s. bf16 WMMA
// (16x16x32) x3-split keeps the 34.4 GFLOP GEMM near that floor; exact
// fp32 WMMA (16x16x4) would be ~8x more matrix instructions.
// X / out are non-temporal (streamed once); packed P stays hot in L2.
// ---------------------------------------------------------------------------

typedef __attribute__((ext_vector_type(16))) __bf16 v16bf;
typedef __attribute__((ext_vector_type(8)))  float  v8f;
typedef __attribute__((ext_vector_type(4)))  float  f4v;   // native vec, NT-load ok
typedef unsigned int uint4v __attribute__((ext_vector_type(4)));

#define N_SAMPLES 16384
#define IN_F      2048
#define NCOMP     512
#define OUTC      2561          // 1 + 512*5
#define MB        32            // rows per block (2 row-tiles of 16)
#define KC        256           // K staged in LDS per iteration
#define LDA       (KC + 8)      // pad to break LDS bank conflicts
#define SQRT_EPS  1e-8f

union BFrag {                   // 16 bf16 = 8 VGPRs, loaded as 2x b128
    v16bf  v;
    uint4v q[2];
};

static __device__ __forceinline__ __bf16 f2bf(float x) { return (__bf16)x; }

// ---------------------------------------------------------------------------
// Pre-split projection P (fp32) into hi/lo bf16 in the exact per-lane
// fragment layout used here for V_WMMA_F32_16X16X32_BF16's B operand, contiguous-K (32x16, lane=N
// column + K-half, 16 contiguous elements per lane => one 32B load/lane).
// Element index: (((k/32)*32 + coltile)*512 + lane*16 + e)
//   k   = (k/32)*32 + (lane<16 ? 0 : 16) + e
//   col = coltile*16 + (lane & 15)
// ---------------------------------------------------------------------------
__global__ void prepack_b(const float* __restrict__ proj,
                          __bf16* __restrict__ bhi,
                          __bf16* __restrict__ blo) {
    int idx  = blockIdx.x * 256 + threadIdx.x;   // 0 .. 2^20-1
    int e    = idx & 15;
    int lane = (idx >> 4) & 31;
    int tile = idx >> 9;                         // 0..2047
    int ctg  = tile & 31;                        // column tile 0..31
    int kc2  = tile >> 5;                        // K/32 chunk 0..63
    int k    = kc2 * 32 + ((lane & 16) ? 16 : 0) + e;
    int col  = ctg * 16 + (lane & 15);
    float x  = proj[(size_t)k * NCOMP + col];
    __bf16 h = f2bf(x);
    __bf16 l = f2bf(x - (float)h);
    bhi[idx] = h;
    blo[idx] = l;
}

// ---------------------------------------------------------------------------
// Main fused kernel. Grid: N_SAMPLES/MB = 512 blocks x 256 threads (8 waves).
// Wave w owns columns [w*64, w*64+64): 2 row-tiles x 4 col-tiles of 16x16.
// A-chunk staging is double-buffered through registers: the next chunk's
// global loads are issued before the WMMA loop so their latency hides
// behind 192 WMMAs.
// ---------------------------------------------------------------------------
__global__ __launch_bounds__(256)
void poly_wmma(const float* __restrict__ X,
               const __bf16* __restrict__ bhi,
               const __bf16* __restrict__ blo,
               float* __restrict__ out) {
    __shared__ __bf16 sAhi[MB * LDA];
    __shared__ __bf16 sAlo[MB * LDA];
    __shared__ float  sXp[MB * 3];     // Xp columns 0..2 for the cross terms

    const int tid    = threadIdx.x;
    const int lane   = tid & 31;
    const int wave   = tid >> 5;
    const int r0     = blockIdx.x * MB;
    const int arow   = lane & 15;             // A-fragment row within tile
    const int hiHalf = (lane >> 4) & 1;       // lane group (0..15 vs 16..31)

    v8f acc[2][4];
    #pragma unroll
    for (int rt = 0; rt < 2; ++rt)
        #pragma unroll
        for (int ct = 0; ct < 4; ++ct)
            acc[rt][ct] = v8f{};

    // ---- prefetch first A chunk (non-temporal: X is streamed once) ----
    f4v xr[8];
    #pragma unroll
    for (int i = 0; i < 8; ++i) {
        int f   = tid + i * 256;
        int row = f >> 6;
        int c4  = f & 63;
        xr[i] = __builtin_nontemporal_load(
            (const f4v*)(X + (size_t)(r0 + row) * IN_F + c4 * 4));
    }

    for (int kc = 0; kc < IN_F; kc += KC) {
        __syncthreads();                       // LDS free from previous chunk
        // ---- split staged fp32 -> (hi,lo) bf16 into LDS ----
        #pragma unroll
        for (int i = 0; i < 8; ++i) {
            int f   = tid + i * 256;
            int row = f >> 6;
            int c4  = f & 63;
            f4v x = xr[i];
            __bf16 h0 = f2bf(x.x), h1 = f2bf(x.y), h2 = f2bf(x.z), h3 = f2bf(x.w);
            __bf16 l0 = f2bf(x.x - (float)h0), l1 = f2bf(x.y - (float)h1);
            __bf16 l2 = f2bf(x.z - (float)h2), l3 = f2bf(x.w - (float)h3);
            __bf16* ph = &sAhi[row * LDA + c4 * 4];
            __bf16* pl = &sAlo[row * LDA + c4 * 4];
            ph[0] = h0; ph[1] = h1; ph[2] = h2; ph[3] = h3;
            pl[0] = l0; pl[1] = l1; pl[2] = l2; pl[3] = l3;
        }
        // ---- issue next chunk's global loads (hidden behind WMMA loop) ----
        if (kc + KC < IN_F) {
            #pragma unroll
            for (int i = 0; i < 8; ++i) {
                int f   = tid + i * 256;
                int row = f >> 6;
                int c4  = f & 63;
                xr[i] = __builtin_nontemporal_load(
                    (const f4v*)(X + (size_t)(r0 + row) * IN_F +
                                 (kc + KC) + c4 * 4));
            }
        }
        __syncthreads();                       // LDS chunk ready

        // ---- WMMA inner loop over K in steps of 32 ----
        #pragma unroll
        for (int k2 = 0; k2 < KC; k2 += 32) {
            // A fragments (contiguous K order, matching packed B):
            //   lanes 0-15 : K = k2+[0..15]
            //   lanes16-31 : K = k2+[16..31]
            BFrag ahi[2], alo[2];
            #pragma unroll
            for (int rt = 0; rt < 2; ++rt) {
                int row = rt * 16 + arow;
                int kb  = k2 + hiHalf * 16;
                const __bf16* bh = &sAhi[row * LDA + kb];
                const __bf16* bl = &sAlo[row * LDA + kb];
                ahi[rt].q[0] = *(const uint4v*)(bh);
                ahi[rt].q[1] = *(const uint4v*)(bh + 8);
                alo[rt].q[0] = *(const uint4v*)(bl);
                alo[rt].q[1] = *(const uint4v*)(bl + 8);
            }
            const int kglob = kc + k2;
            #pragma unroll
            for (int ct = 0; ct < 4; ++ct) {
                int    ctg  = wave * 4 + ct;
                size_t boff = ((size_t)(kglob >> 5) * 32 + ctg) * 512 + lane * 16;
                BFrag bh, bl;
                bh.q[0] = *(const uint4v*)(bhi + boff);
                bh.q[1] = *(const uint4v*)(bhi + boff + 8);
                bl.q[0] = *(const uint4v*)(blo + boff);
                bl.q[1] = *(const uint4v*)(blo + boff + 8);
                #pragma unroll
                for (int rt = 0; rt < 2; ++rt) {
                    // hi*hi + hi*lo + lo*hi  (bf16x3 split, fp32 accumulate)
                    acc[rt][ct] = __builtin_amdgcn_wmma_f32_16x16x32_bf16(
                        false, ahi[rt].v, false, bh.v, (short)0, acc[rt][ct],
                        false, false);
                    acc[rt][ct] = __builtin_amdgcn_wmma_f32_16x16x32_bf16(
                        false, ahi[rt].v, false, bl.v, (short)0, acc[rt][ct],
                        false, false);
                    acc[rt][ct] = __builtin_amdgcn_wmma_f32_16x16x32_bf16(
                        false, alo[rt].v, false, bh.v, (short)0, acc[rt][ct],
                        false, false);
                }
            }
        }
    }

    // ---- share Xp columns 0..2 across block (cross terms need them) ----
    // C/D layout: VGPR g, lanes0-15 -> M=g, lanes16-31 -> M=8+g; N = lane&15.
    if (wave == 0 && (lane & 15) < 3) {
        int col = lane & 15;
        #pragma unroll
        for (int rt = 0; rt < 2; ++rt)
            #pragma unroll
            for (int g = 0; g < 8; ++g)
                sXp[(rt * 16 + hiHalf * 8 + g) * 3 + col] = acc[rt][0][g];
    }
    __syncthreads();

    // ---- fused epilogue: bias | p1 | p2 | p3 | cross | sqrt ----
    // All stores non-temporal: output (168 MB) is write-only, keep L2 for P.
    #pragma unroll
    for (int rt = 0; rt < 2; ++rt) {
        int rlb = rt * 16 + hiHalf * 8;
        #pragma unroll
        for (int ct = 0; ct < 4; ++ct) {
            int col = wave * 64 + ct * 16 + (lane & 15);
            #pragma unroll
            for (int g = 0; g < 8; ++g) {
                int    rl   = rlb + g;
                size_t base = (size_t)(r0 + rl) * OUTC;
                float  v    = acc[rt][ct][g];
                float  v2   = v * v;
                __builtin_nontemporal_store(v,      out + base + 1    + col);
                __builtin_nontemporal_store(v2,     out + base + 513  + col);
                __builtin_nontemporal_store(v2 * v, out + base + 1025 + col);
                __builtin_nontemporal_store(sqrtf(fabsf(v) + SQRT_EPS),
                                            out + base + 2049 + col);
                if (col > 0) {
                    // pairs[:, col-1] = Xp[:,0] * Xp[:,col]
                    __builtin_nontemporal_store(sXp[rl * 3 + 0] * v,
                                                out + base + 1537 + (col - 1));
                } else {
                    __builtin_nontemporal_store(1.0f, out + base + 0); // bias
                    // pairs[:, 511] = Xp[:,1] * Xp[:,2]
                    __builtin_nontemporal_store(
                        sXp[rl * 3 + 1] * sXp[rl * 3 + 2],
                        out + base + 1537 + 511);
                }
            }
        }
    }
}

// ---------------------------------------------------------------------------
extern "C" void kernel_launch(void* const* d_in, const int* in_sizes, int n_in,
                              void* d_out, int out_size, void* d_ws, size_t ws_size,
                              hipStream_t stream) {
    const float* X    = (const float*)d_in[0];   // [16384, 2048] fp32
    const float* proj = (const float*)d_in[1];   // [2048, 512]  fp32
    float* out = (float*)d_out;                  // [16384, 2561] fp32

    // Workspace: split-bf16 packed projection (2 MB hi + 2 MB lo)
    __bf16* bhi = (__bf16*)d_ws;
    __bf16* blo = bhi + (size_t)IN_F * NCOMP;

    prepack_b<<<(IN_F * NCOMP) / 256, 256, 0, stream>>>(proj, bhi, blo);
    poly_wmma<<<N_SAMPLES / MB, 256, 0, stream>>>(X, bhi, blo, out);
}
